// Critic_41162966565055
// MI455X (gfx1250) — matrix operand order repaired and checked by reading, not compiled, not run on hardware
//
#include <hip/hip_runtime.h>
#include <hip/hip_bf16.h>
#include <math.h>

// ---------------------------------------------------------------------------
// Sizes from the reference
// ---------------------------------------------------------------------------
#define B_SZ    512
#define T_SZ    100
#define F_MS    64
#define F_RS    128
#define F_ACT   128
#define F_OSC   256
#define U_MS    256
#define U_RS    256
#define U_COMB  512
#define U_LSTM  512
#define OSC_HALF 128
#define LSTM_IN 640      // 128 + 256 + 256
#define KTOT    1152     // LSTM_IN + U_LSTM (combined K for x@Wk + h@Wrk)
#define NGATE4  2048     // 4 * U_LSTM
#define NSTEPS  199      // 99 warmup + 100 main
#define LDS_STRIDE 1160  // 1152 + 8 bf16 pad -> conflict-reduced ds_load_b128

typedef __attribute__((ext_vector_type(16))) __bf16 v16bf;
typedef __attribute__((ext_vector_type(8)))  __bf16 v8bf;
typedef __attribute__((ext_vector_type(4)))  __bf16 v4bf;
typedef __attribute__((ext_vector_type(8)))  float  v8f;
typedef __attribute__((ext_vector_type(4)))  float  v4f;

__device__ __forceinline__ float elu1(float x)     { return x > 0.f ? x : expm1f(x); }
__device__ __forceinline__ float sigmoidf_(float x){ return 1.f / (1.f + __expf(-x)); }

// ---------------------------------------------------------------------------
// Stage 1: small dense layers (off critical path, naive fp32)
// ---------------------------------------------------------------------------
__global__ void k_dense_elu(const float* __restrict__ in, const float* __restrict__ W,
                            const float* __restrict__ b, float* __restrict__ out,
                            int K, int N) {
    int idx = blockIdx.x * blockDim.x + threadIdx.x;  // B*N threads, exact
    int bi = idx / N, j = idx % N;
    const float* row = in + (size_t)bi * K;
    float acc = b[j];
    for (int k = 0; k < K; ++k) acc += row[k] * W[(size_t)k * N + j];
    out[idx] = elu1(acc);
}

__global__ void k_state(const float* __restrict__ ms, const float* __restrict__ rs,
                        const float* __restrict__ Wc, const float* __restrict__ bc,
                        float* __restrict__ h0, float* __restrict__ c0) {
    int idx = blockIdx.x * blockDim.x + threadIdx.x;  // B*U_COMB threads
    int bi = idx / U_COMB, j = idx % U_COMB;
    float acc = bc[j];
    const float* mrow = ms + (size_t)bi * U_MS;
    const float* rrow = rs + (size_t)bi * U_RS;
    for (int k = 0; k < U_MS; ++k) acc += mrow[k] * Wc[(size_t)k * U_COMB + j];
    for (int k = 0; k < U_RS; ++k) acc += rrow[k] * Wc[(size_t)(U_MS + k) * U_COMB + j];
    float v = elu1(acc);
    h0[idx] = v;
    c0[idx] = v;
}

// ---------------------------------------------------------------------------
// Weight prep: WzT[n][k] = bf16( k<640 ? Wk[k][n] : Wrk[k-640][n] )
// 4.7 MB bf16 -> resident in the 192 MB L2 for all 199 steps.
// ---------------------------------------------------------------------------
__global__ void k_wzt(const float* __restrict__ Wk, const float* __restrict__ Wrk,
                      __bf16* __restrict__ WzT) {
    int idx = blockIdx.x * blockDim.x + threadIdx.x;  // NGATE4*KTOT threads
    int n = idx / KTOT, k = idx % KTOT;
    float v = (k < LSTM_IN) ? Wk[(size_t)k * NGATE4 + n]
                            : Wrk[(size_t)(k - LSTM_IN) * NGATE4 + n];
    WzT[idx] = (__bf16)v;
}

// ---------------------------------------------------------------------------
// Stage 2 (parallel over all 199 steps): build X_all bf16 [199,512,640]
// ---------------------------------------------------------------------------
__global__ void k_copy1(const float* __restrict__ action, const float* __restrict__ history,
                        __bf16* __restrict__ X) {
    size_t idx = (size_t)blockIdx.x * blockDim.x + threadIdx.x;  // 199*512*128
    int j = (int)(idx % F_ACT);
    size_t r = idx / F_ACT;
    int b = (int)(r % B_SZ);
    int t = (int)(r / B_SZ);
    float v = (t < 99) ? history[((size_t)b * T_SZ + (t + 1)) * F_ACT + j]
                       : action [((size_t)b * T_SZ + (t - 99)) * F_ACT + j];
    X[((size_t)t * B_SZ + b) * LSTM_IN + j] = (__bf16)v;
}

__global__ void k_inp2(const float* __restrict__ osc, const float* __restrict__ hosc,
                       const float* __restrict__ Wor, const float* __restrict__ bor,
                       __bf16* __restrict__ X) {
    size_t idx = (size_t)blockIdx.x * blockDim.x + threadIdx.x;  // 199*512*256
    int j = (int)(idx % 256);
    size_t r = idx / 256;
    int b = (int)(r % B_SZ);
    int t = (int)(r / B_SZ);
    const float* orow = (t < 99) ? (hosc + ((size_t)b * T_SZ + (t + 1)) * F_OSC)
                                 : (osc  + ((size_t)b * T_SZ + (t - 99)) * F_OSC);
    float acc = bor[j];
    for (int k = 0; k < OSC_HALF; ++k) acc += orow[k] * Wor[(size_t)k * 256 + j];
    X[((size_t)t * B_SZ + b) * LSTM_IN + F_ACT + j] = (__bf16)elu1(acc);
}

__global__ void k_inp3(const float* __restrict__ Woi, const float* __restrict__ boi,
                       __bf16* __restrict__ X) {
    size_t idx = (size_t)blockIdx.x * blockDim.x + threadIdx.x;  // 199*512*256
    int j = (int)(idx % 256);
    size_t r = idx / 256;
    int b = (int)(r % B_SZ);
    int t = (int)(r / B_SZ);
    const __bf16* x2 = X + ((size_t)t * B_SZ + b) * LSTM_IN + (F_ACT + 128); // inp2[:,128:]
    float acc = boi[j];
    for (int k = 0; k < 128; ++k) acc += (float)x2[k] * Woi[(size_t)k * 256 + j];
    X[((size_t)t * B_SZ + b) * LSTM_IN + 384 + j] = (__bf16)elu1(acc);
}

// ---------------------------------------------------------------------------
// Sequential LSTM step (critical path), fused WMMA bf16->f32.
//
//  - Block = 128 threads = 4 waves.  Block tile: 32 rows x (4 units x 16).
//  - A tile (32 rows x 1152 K: X bf16 + h cvt'd to bf16) staged ONCE in LDS,
//    shared by all 4 waves (4x less A traffic, one fp32->bf16 pass).
//  - Each wave: 2 m-tiles x 4 gate accumulators (8 x v8f) so every streamed
//    B fragment feeds two WMMAs -> WzT L2 traffic halves vs 1 m-tile/wave.
//  - B fragments double-buffered in registers (load k+32 while computing k)
//    to kill the load->wmma s_wait_loadcnt stalls.
// Grid: 16 m-groups x 8 u-groups = 128 blocks.
// ---------------------------------------------------------------------------
__global__ __launch_bounds__(128) void k_step(
    const __bf16* __restrict__ X,    // [512,640]  this step's input fragment
    const __bf16* __restrict__ WzT,  // [2048,1152] transposed combined weights
    const float*  __restrict__ bl,   // [2048]
    const float*  __restrict__ hin,  // [512,512]
    float*        __restrict__ hout, // [512,512]
    float*        __restrict__ c)    // [512,512]
{
    __shared__ __bf16 As[32][LDS_STRIDE];   // 74,240 B of the 320 KB/WGP

    const int tid  = threadIdx.x;
    const int wave = tid >> 5;
    const int lane = tid & 31;
    const int m0 = (blockIdx.x >> 3) * 32;              // 16 m-groups
    const int n0 = ((blockIdx.x & 7) * 4 + wave) * 16;  // 8 u-groups x 4 waves
    const int lg = lane & 15;   // N col / M row selector
    const int lh = lane >> 4;   // half-wave selector

    // ---- stage A tile in LDS (cooperative, 128 threads) ----
    // X part: bf16 straight copy, 32 rows x 640
    for (int i = tid; i < (32 * LSTM_IN) / 8; i += 128) {
        int row = i / (LSTM_IN / 8);
        int col = (i % (LSTM_IN / 8)) * 8;
        *(v8bf*)&As[row][col] = *(const v8bf*)(X + (size_t)(m0 + row) * LSTM_IN + col);
    }
    // h part: fp32 -> bf16 once, 32 rows x 512 at K offset 640
    for (int i = tid; i < (32 * U_LSTM) / 4; i += 128) {
        int row = i / (U_LSTM / 4);
        int col = (i % (U_LSTM / 4)) * 4;
        v4f hv = *(const v4f*)(hin + (size_t)(m0 + row) * U_LSTM + col);
        v4bf o;
#pragma unroll
        for (int j = 0; j < 4; ++j) o[j] = (__bf16)hv[j];
        *(v4bf*)&As[row][LSTM_IN + col] = o;
    }
    __syncthreads();

    // ---- per-gate B base pointers (row = gate*512 + unit column) ----
    const __bf16* Bbase[4];
#pragma unroll
    for (int g = 0; g < 4; ++g)
        Bbase[g] = WzT + (size_t)(g * U_LSTM + n0 + lg) * KTOT + lh * 16;

    v8f acc[2][4];
#pragma unroll
    for (int mt = 0; mt < 2; ++mt)
#pragma unroll
        for (int g = 0; g < 4; ++g)
#pragma unroll
            for (int i = 0; i < 8; ++i) acc[mt][g][i] = 0.0f;

    // B fragment loader (contiguous k order, same as loadA: lane group picks K half,
    // 16 contiguous K per lane)
    auto loadB = [&](int k, v16bf (&bf)[4]) {
#pragma unroll
        for (int g = 0; g < 4; ++g) {
            v8bf b0 = *(const v8bf*)(Bbase[g] + k);
            v8bf b1 = *(const v8bf*)(Bbase[g] + k + 8);
#pragma unroll
            for (int i = 0; i < 8; ++i) { bf[g][i] = b0[i]; bf[g][8 + i] = b1[i]; }
        }
    };
    // A fragment from LDS (contiguous k order, same as loadB: lane<16 K{0..15},
    // lane>=16 K{16..31})
    auto loadA = [&](int k, int mt) -> v16bf {
        const __bf16* p = &As[mt * 16 + lg][k + lh * 16];
        v8bf a0 = *(const v8bf*)p;
        v8bf a1 = *(const v8bf*)(p + 8);
        v16bf a;
#pragma unroll
        for (int i = 0; i < 8; ++i) { a[i] = a0[i]; a[8 + i] = a1[i]; }
        return a;
    };

    // ---- main K loop, unrolled x2 with register double-buffered B ----
    v16bf bA[4], bB[4];
    loadB(0, bA);
    for (int k0 = 0; k0 < KTOT; k0 += 64) {
        loadB(k0 + 32, bB);                       // prefetch next chunk
        {
            v16bf a0 = loadA(k0, 0);
            v16bf a1 = loadA(k0, 1);
#pragma unroll
            for (int g = 0; g < 4; ++g)
                acc[0][g] = __builtin_amdgcn_wmma_f32_16x16x32_bf16(
                    false, a0, false, bA[g], (short)0, acc[0][g], false, false);
#pragma unroll
            for (int g = 0; g < 4; ++g)
                acc[1][g] = __builtin_amdgcn_wmma_f32_16x16x32_bf16(
                    false, a1, false, bA[g], (short)0, acc[1][g], false, false);
        }
        if (k0 + 64 < KTOT) loadB(k0 + 64, bA);   // prefetch chunk after next
        {
            v16bf a0 = loadA(k0 + 32, 0);
            v16bf a1 = loadA(k0 + 32, 1);
#pragma unroll
            for (int g = 0; g < 4; ++g)
                acc[0][g] = __builtin_amdgcn_wmma_f32_16x16x32_bf16(
                    false, a0, false, bB[g], (short)0, acc[0][g], false, false);
#pragma unroll
            for (int g = 0; g < 4; ++g)
                acc[1][g] = __builtin_amdgcn_wmma_f32_16x16x32_bf16(
                    false, a1, false, bB[g], (short)0, acc[1][g], false, false);
        }
    }

    // ---- fused LSTM cell update (keras gate order i,f,g,o) ----
    // C/D layout: element e of a lane -> M = e + 8*lh, N = lg (lane-invariant)
    const int u = n0 + lg;
    const float bi_ = bl[u];
    const float bf_ = bl[512 + u];
    const float bg_ = bl[1024 + u];
    const float bo_ = bl[1536 + u];
#pragma unroll
    for (int mt = 0; mt < 2; ++mt) {
#pragma unroll
        for (int e = 0; e < 8; ++e) {
            int m = m0 + mt * 16 + e + lh * 8;
            size_t off = (size_t)m * U_LSTM + u;
            float zi = acc[mt][0][e] + bi_;
            float zf = acc[mt][1][e] + bf_;
            float zg = acc[mt][2][e] + bg_;
            float zo = acc[mt][3][e] + bo_;
            float cn = sigmoidf_(zf) * c[off] + sigmoidf_(zi) * tanhf(zg);
            float hn = sigmoidf_(zo) * tanhf(cn);
            c[off]    = cn;
            hout[off] = hn;
        }
    }
}

// ---------------------------------------------------------------------------
// Output head: out[b] = elu(h @ Wo + bo), U_OUT = 1
// ---------------------------------------------------------------------------
__global__ void k_out(const float* __restrict__ h, const float* __restrict__ Wo,
                      const float* __restrict__ bo, float* __restrict__ out) {
    int b = blockIdx.x * blockDim.x + threadIdx.x;
    if (b >= B_SZ) return;
    float acc = bo[0];
    const float* hrow = h + (size_t)b * U_LSTM;
    for (int k = 0; k < U_LSTM; ++k) acc += hrow[k] * Wo[k];
    out[b] = elu1(acc);
}

// ---------------------------------------------------------------------------
// Host side
// ---------------------------------------------------------------------------
extern "C" void kernel_launch(void* const* d_in, const int* in_sizes, int n_in,
                              void* d_out, int out_size, void* d_ws, size_t ws_size,
                              hipStream_t stream) {
    (void)in_sizes; (void)n_in; (void)out_size; (void)ws_size;

    const float* motion  = (const float*)d_in[0];
    const float* robot   = (const float*)d_in[1];
    /* d_in[2] = osc_state, unused by the reference call path */
    const float* action  = (const float*)d_in[3];
    const float* osc     = (const float*)d_in[4];
    const float* history = (const float*)d_in[5];
    const float* hosc    = (const float*)d_in[6];
    const float* Wm  = (const float*)d_in[7];
    const float* bm  = (const float*)d_in[8];
    const float* Wr  = (const float*)d_in[9];
    const float* br  = (const float*)d_in[10];
    const float* Wc  = (const float*)d_in[11];
    const float* bc  = (const float*)d_in[12];
    const float* Wor = (const float*)d_in[13];
    const float* bor = (const float*)d_in[14];
    const float* Woi = (const float*)d_in[15];
    const float* boi = (const float*)d_in[16];
    const float* Wk  = (const float*)d_in[17];
    const float* Wrk = (const float*)d_in[18];
    const float* bl  = (const float*)d_in[19];
    const float* Wo  = (const float*)d_in[20];
    const float* bo  = (const float*)d_in[21];
    float* out = (float*)d_out;

    // workspace carve-out (256 B aligned):
    //   X_all bf16 [199,512,640] ~130.4 MB | WzT bf16 [2048,1152] ~4.7 MB
    //   ms,rs f32 | h0,h1 f32 ping-pong | c f32
    char* ws = (char*)d_ws;
    size_t off = 0;
    auto carve = [&](size_t bytes) -> char* {
        off = (off + 255) & ~(size_t)255;
        char* p = ws + off;
        off += bytes;
        return p;
    };
    __bf16* Xall = (__bf16*)carve((size_t)NSTEPS * B_SZ * LSTM_IN * sizeof(__bf16));
    __bf16* WzT  = (__bf16*)carve((size_t)NGATE4 * KTOT * sizeof(__bf16));
    float* msbuf = (float*)carve((size_t)B_SZ * U_MS * sizeof(float));
    float* rsbuf = (float*)carve((size_t)B_SZ * U_RS * sizeof(float));
    float* hbuf0 = (float*)carve((size_t)B_SZ * U_LSTM * sizeof(float));
    float* hbuf1 = (float*)carve((size_t)B_SZ * U_LSTM * sizeof(float));
    float* cbuf  = (float*)carve((size_t)B_SZ * U_LSTM * sizeof(float));

    // --- weight prep + stage 1 ---
    k_wzt<<<(NGATE4 * KTOT) / 256, 256, 0, stream>>>(Wk, Wrk, WzT);
    k_dense_elu<<<(B_SZ * U_MS) / 256, 256, 0, stream>>>(motion, Wm, bm, msbuf, F_MS, U_MS);
    k_dense_elu<<<(B_SZ * U_RS) / 256, 256, 0, stream>>>(robot, Wr, br, rsbuf, F_RS, U_RS);
    k_state<<<(B_SZ * U_COMB) / 256, 256, 0, stream>>>(msbuf, rsbuf, Wc, bc, hbuf0, cbuf);

    // --- stage 2: build all 199 step inputs in parallel ---
    k_copy1<<<(NSTEPS * B_SZ * F_ACT) / 256, 256, 0, stream>>>(action, history, Xall);
    k_inp2 <<<(NSTEPS * B_SZ * 256) / 256, 256, 0, stream>>>(osc, hosc, Wor, bor, Xall);
    k_inp3 <<<(NSTEPS * B_SZ * 256) / 256, 256, 0, stream>>>(Woi, boi, Xall);

    // --- sequential scan: 199 fused WMMA steps, h ping-pong ---
    for (int t = 0; t < NSTEPS; ++t) {
        const float* hin  = (t & 1) ? hbuf1 : hbuf0;
        float*       hout = (t & 1) ? hbuf0 : hbuf1;
        k_step<<<128, 128, 0, stream>>>(Xall + (size_t)t * B_SZ * LSTM_IN,
                                        WzT, bl, hin, hout, cbuf);
    }
    // NSTEPS = 199, last step t=198 (even) wrote hbuf1
    k_out<<<2, 256, 0, stream>>>(hbuf1, Wo, bo, out);
}
